// myAdvPTRNNModel_47407849013455
// MI455X (gfx1250) — matrix operand order repaired and checked by reading, not compiled, not run on hardware
//
#include <hip/hip_runtime.h>

#define BATCH 512
#define SEQ   1024
#define HID   64
#define EMB   32
#define VOCAB 10
#define NOUT  10

typedef __attribute__((ext_vector_type(16))) _Float16 v16h;
typedef __attribute__((ext_vector_type(8)))  _Float16 v8h;
typedef __attribute__((ext_vector_type(8)))  float    v8f;

#define MROWS 16      // batch rows per block
#define HSTR  72      // padded LDS row stride in halves (144B -> conflict-free A loads)

struct Smem {
  unsigned short pk[SEQ * MROWS];              // step-major: pk[s*16+r] = num1 | num2<<8 (32 KB)
  alignas(16) _Float16 emb[VOCAB * EMB];       // f16 embedding table (rows are 64B)
  alignas(16) _Float16 h1buf[2][MROWS * HSTR]; // ping-pong layer-1 state (f16)
  alignas(16) _Float16 h2buf[2][MROWS * HSTR]; // ping-pong layer-2 state (f16)
};

// tanh(x) = 1 - 2/(1 + e^{2x}); exact at both saturations (t->0 => -1, t->inf => +1).
__device__ __forceinline__ float fast_tanh(float x) {
  float t = __builtin_amdgcn_exp2f(x * 2.885390081777927f);   // e^{2x}
  return __builtin_fmaf(-2.0f, __builtin_amdgcn_rcpf(1.0f + t), 1.0f);
}

// A operand (16x32 f16 tile at K-block k0) from a padded-stride state buffer.
// lane<16: row=lane, K=k0+[0..7],[16..23]; lane>=16: row=lane-16, K=k0+[8..15],[24..31].
__device__ __forceinline__ v16h loadA(const _Float16* buf, int lane, int k0) {
  int r  = lane & 15;
  int kc = k0 + ((lane >> 4) << 3);
  v8h lo = *(const v8h*)(buf + r * HSTR + kc);
  v8h hi = *(const v8h*)(buf + r * HSTR + kc + 16);
  v16h a;
#pragma unroll
  for (int j = 0; j < 8; ++j) { a[j] = lo[j]; a[8 + j] = hi[j]; }
  return a;
}

// A operand built directly from one 32-wide embedding row (row chosen per lane's M row).
__device__ __forceinline__ v16h loadAemb(const _Float16* embrow, int lane) {
  int kc = (lane >> 4) << 3;                   // 0 or 8
  v8h lo = *(const v8h*)(embrow + kc);
  v8h hi = *(const v8h*)(embrow + kc + 16);
  v16h a;
#pragma unroll
  for (int j = 0; j < 8; ++j) { a[j] = lo[j]; a[8 + j] = hi[j]; }
  return a;
}

// B operand (32x16 f16) for D = A @ W^T: B[k][n] = W[n][k], W row-major [n][64].
// Loaded once from global, f32->f16, lives in VGPRs for the whole kernel.
__device__ __forceinline__ v16h makeB(const float* W, int nrows, int n, int k0, int lane) {
  int kb = k0 + ((lane >> 4) << 3);
  v16h b;
  if (n < nrows) {
    const float* p = W + n * HID + kb;
#pragma unroll
    for (int j = 0; j < 16; ++j) b[j] = (_Float16)p[j < 8 ? j : j + 8];
  } else {
#pragma unroll
    for (int j = 0; j < 16; ++j) b[j] = (_Float16)0.0f;
  }
  return b;
}

#define WMMA(a, b, c) \
  __builtin_amdgcn_wmma_f32_16x16x32_f16(false, (a), false, (b), (short)0, (c), false, false)

__global__ __launch_bounds__(128)
void rnn_fused_kernel(const int* __restrict__ num1, const int* __restrict__ num2,
                      const float* __restrict__ h0, const float* __restrict__ embed_w,
                      const float* __restrict__ w_ih, const float* __restrict__ w_hh,
                      const float* __restrict__ b_ih, const float* __restrict__ b_hh,
                      const float* __restrict__ dense_w, const float* __restrict__ dense_b,
                      float* __restrict__ out) {
  __shared__ Smem sm;
  const int tid   = threadIdx.x;
  const int lane  = tid & 31;
  const int wv    = tid >> 5;       // wave 0..3, owns hidden columns [16*wv, 16*wv+16)
  const int nw    = wv * 16;
  const int o     = lane & 15;      // D-tile column within this wave's tile
  const int sel   = lane >> 4;      // D-tile row-half selector
  const int rr    = lane & 15;      // A-tile row this lane serves
  const int bbase = blockIdx.x * MROWS;
  float* hfin = out + (size_t)BATCH * SEQ * NOUT;

  // ---------------- init (off critical path) ----------------
  for (int idx = tid; idx < MROWS * SEQ; idx += 128) {
    int r = idx >> 10, s = idx & (SEQ - 1);
    int g = (bbase + r) * SEQ + s;
    sm.pk[s * MROWS + r] = (unsigned short)((num1[g] & 0xFF) | ((num2[g] & 0xFF) << 8));
  }
  for (int j = tid; j < VOCAB * EMB; j += 128) sm.emb[j] = (_Float16)embed_w[j];
  {
    int r = tid >> 3, c0 = (tid & 7) * 8;
#pragma unroll
    for (int j = 0; j < 8; ++j) {
      _Float16 a = (_Float16)h0[(bbase + r) * HID + c0 + j];
      _Float16 b = (_Float16)h0[BATCH * HID + (bbase + r) * HID + c0 + j];
      sm.h1buf[0][r * HSTR + c0 + j] = a;
      sm.h2buf[0][r * HSTR + c0 + j] = b;   // h2[-2] slot parity 0
      sm.h2buf[1][r * HSTR + c0 + j] = b;   // h2[-1] slot parity 1 (read at iter 1)
    }
  }

  // Weight B-operands -> persistent VGPRs (f16)
  const int nn = nw + o;
  v16h Bih1_0 = makeB(w_ih,             HID, nn, 0,  lane);
  v16h Bih1_1 = makeB(w_ih,             HID, nn, 32, lane);
  v16h Bhh1_0 = makeB(w_hh,             HID, nn, 0,  lane);
  v16h Bhh1_1 = makeB(w_hh,             HID, nn, 32, lane);
  v16h Bih2_0 = makeB(w_ih + HID * HID, HID, nn, 0,  lane);
  v16h Bih2_1 = makeB(w_ih + HID * HID, HID, nn, 32, lane);
  v16h Bhh2_0 = makeB(w_hh + HID * HID, HID, nn, 0,  lane);
  v16h Bhh2_1 = makeB(w_hh + HID * HID, HID, nn, 32, lane);
  v16h Bd_0 = makeB(dense_w, (wv == 0) ? NOUT : 0, o, 0,  lane);
  v16h Bd_1 = makeB(dense_w, (wv == 0) ? NOUT : 0, o, 32, lane);

  const float bias1 = b_ih[nn] + b_hh[nn];
  const float bias2 = b_ih[HID + nn] + b_hh[HID + nn];
  const float biasd = (o < NOUT) ? dense_b[o] : 0.0f;

  __syncthreads();

  // Prefetched x A-fragments for the upcoming step (recurrence-independent).
  v16h axc0, axc1;
  {
    unsigned pv = sm.pk[rr];
    axc0 = loadAemb(sm.emb + (pv & 0xFF) * EMB, lane);
    axc1 = loadAemb(sm.emb + (pv >> 8)   * EMB, lane);
  }

  // iter i computes h1[i], h2[i-1], logits[i-2].
  // Parity: h1buf[i&1] holds h1[i-1], h2buf[i&1] holds h2[i-2]; writes go to parity^1.
  auto body = [&](int i, bool doL1, bool doL2, bool doD, bool lastL2) __attribute__((always_inline)) {
    const int cur = i & 1, nxt = cur ^ 1;
    v16h ah1_0, ah1_1, ah2_0, ah2_1;
    if (doL1 || doL2) {
      ah1_0 = loadA(sm.h1buf[cur], lane, 0);   // h1[i-1]: L1-recurrent + L2-input
      ah1_1 = loadA(sm.h1buf[cur], lane, 32);
    }
    if (doL2 || doD) {
      ah2_0 = loadA(sm.h2buf[cur], lane, 0);   // h2[i-2]: L2-recurrent + dense
      ah2_1 = loadA(sm.h2buf[cur], lane, 32);
    }
    v8f acc1, acc2;
    if (doL1) {
#pragma unroll
      for (int v = 0; v < 8; ++v) acc1[v] = bias1;
      acc1 = WMMA(axc0,  Bih1_0, acc1);        // prefetched operands: no LDS wait
      acc1 = WMMA(axc1,  Bih1_1, acc1);
      acc1 = WMMA(ah1_0, Bhh1_0, acc1);
      acc1 = WMMA(ah1_1, Bhh1_1, acc1);
      {  // prefetch x for step i+1 in the shadow of the WMMAs/tanh
        int sp = (i + 1 < SEQ) ? (i + 1) : (SEQ - 1);
        unsigned pv = sm.pk[sp * MROWS + rr];
        axc0 = loadAemb(sm.emb + (pv & 0xFF) * EMB, lane);
        axc1 = loadAemb(sm.emb + (pv >> 8)   * EMB, lane);
      }
#pragma unroll
      for (int v = 0; v < 8; ++v) acc1[v] = fast_tanh(acc1[v]);
    }
    if (doL2) {
#pragma unroll
      for (int v = 0; v < 8; ++v) acc2[v] = bias2;
      acc2 = WMMA(ah1_0, Bih2_0, acc2);
      acc2 = WMMA(ah1_1, Bih2_1, acc2);
      acc2 = WMMA(ah2_0, Bhh2_0, acc2);
      acc2 = WMMA(ah2_1, Bhh2_1, acc2);
#pragma unroll
      for (int v = 0; v < 8; ++v) acc2[v] = fast_tanh(acc2[v]);
    }
    if (doD && wv == 0) {                      // dense head for step s = i-2
      v8f accd;
#pragma unroll
      for (int v = 0; v < 8; ++v) accd[v] = biasd;
      accd = WMMA(ah2_0, Bd_0, accd);
      accd = WMMA(ah2_1, Bd_1, accd);
      if (o < NOUT) {
        const int s = i - 2;
#pragma unroll
        for (int v = 0; v < 8; ++v)
          out[(size_t)((bbase + v + 8 * sel) * SEQ + s) * NOUT + o] = accd[v];
      }
    }
    if (doL1) {
#pragma unroll
      for (int v = 0; v < 8; ++v)
        sm.h1buf[nxt][(v + 8 * sel) * HSTR + nw + o] = (_Float16)acc1[v];
      if (i == SEQ - 1) {                      // final layer-1 hidden (f32)
#pragma unroll
        for (int v = 0; v < 8; ++v)
          hfin[(size_t)(bbase + v + 8 * sel) * HID + nw + o] = acc1[v];
      }
    }
    if (doL2) {
#pragma unroll
      for (int v = 0; v < 8; ++v)
        sm.h2buf[nxt][(v + 8 * sel) * HSTR + nw + o] = (_Float16)acc2[v];
      if (lastL2) {                            // final layer-2 hidden (f32)
#pragma unroll
        for (int v = 0; v < 8; ++v)
          hfin[(size_t)BATCH * HID + (size_t)(bbase + v + 8 * sel) * HID + nw + o] = acc2[v];
      }
    }
    __syncthreads();
  };

  body(0, true, false, false, false);          // h1[0]
  body(1, true, true,  false, false);          // h1[1], h2[0]
  for (int i = 2; i < SEQ; ++i)                // branch-free steady state
    body(i, true, true, true, false);
  body(SEQ,     false, true,  true, true);     // h2[SEQ-1], logits[SEQ-2]
  body(SEQ + 1, false, false, true, false);    // logits[SEQ-1]
}

extern "C" void kernel_launch(void* const* d_in, const int* in_sizes, int n_in,
                              void* d_out, int out_size, void* d_ws, size_t ws_size,
                              hipStream_t stream) {
  const int*   num1    = (const int*)d_in[0];
  const int*   num2    = (const int*)d_in[1];
  const float* h0      = (const float*)d_in[2];
  const float* embed_w = (const float*)d_in[3];
  const float* w_ih    = (const float*)d_in[4];
  const float* w_hh    = (const float*)d_in[5];
  const float* b_ih    = (const float*)d_in[6];
  const float* b_hh    = (const float*)d_in[7];
  const float* dense_w = (const float*)d_in[8];
  const float* dense_b = (const float*)d_in[9];
  rnn_fused_kernel<<<BATCH / MROWS, 128, 0, stream>>>(
      num1, num2, h0, embed_w, w_ih, w_hh, b_ih, b_hh, dense_w, dense_b, (float*)d_out);
}
